// LocalAtten_1666447311313
// MI455X (gfx1250) — compile-verified, matrix operand order checked by reading, not run on hardware
//
#include <hip/hip_runtime.h>
#include <hip/hip_bf16.h>

// ---------------------------------------------------------------------------
// LocalAttention fused pipeline for gfx1250 (MI455X), bf16 WMMA everywhere.
//   Kernel 1: conv1x1 (256->16) + conv1x1 (256->9) + relu + softmax
//             -> ws_xr   [pixel][16] bf16   (pixel-major for the 3x3 gather)
//             -> ws_attn [pixel][16] bf16   (9 softmax weights, padded)
//   Kernel 2: 3x3 attn-weighted gather of xr -> v[pixel][16]
//             conv1x1 (16->256) + bias + relu + residual add of x.
// Grid for both: one block (256 thr = 8 waves) per image row (16*128 = 2048).
// All bulk global traffic is b128-vectorized (HBM-bound op: ~536 MB minimum).
// ---------------------------------------------------------------------------

typedef __bf16 bf16;
typedef __attribute__((ext_vector_type(16))) __bf16 v16bf;
typedef __attribute__((ext_vector_type(8)))  __bf16 v8bf;
typedef __attribute__((ext_vector_type(8)))  float  v8f;

constexpr int kC   = 256;   // channels
constexpr int kCI  = 16;    // inter channels
constexpr int kH   = 128;
constexpr int kW   = 128;
constexpr int kHW  = kH * kW;
constexpr int kN   = 16;    // batch

__device__ __forceinline__ bf16 f2bf(float f) {
  unsigned u = __float_as_uint(f);
  u += 0x7FFFu + ((u >> 16) & 1u);                 // round to nearest even
  unsigned short s = (unsigned short)(u >> 16);
  bf16 h; __builtin_memcpy(&h, &s, 2); return h;
}
__device__ __forceinline__ float bf2f(bf16 h) {
  unsigned short s; __builtin_memcpy(&s, &h, 2);
  return __uint_as_float(((unsigned)s) << 16);
}

// ------------------------------ Kernel 1 -----------------------------------
// LDS strides in halfs, padded so 16-lane ds_load_b128 fragments are
// bank-conflict free (stride*2 bytes ≡ 4 dwords mod 64 -> distinct banks)
constexpr int XSTR = 264;   // xbuf row stride (528B: 132 dw ≡ 4 mod 64)
constexpr int WSTR = 264;

__global__ __launch_bounds__(256, 1)
void k1_conv12(const float* __restrict__ x,
               const float* __restrict__ w1, const float* __restrict__ b1,
               const float* __restrict__ w2, const float* __restrict__ b2,
               bf16* __restrict__ ws_xr, bf16* __restrict__ ws_attn)
{
  __shared__ bf16  xbuf[128 * XSTR];     // [pixel][channel] bf16 A-operand
  __shared__ bf16  wbuf[32 * WSTR];      // rows 0-15: w1, 16-24: w2, rest 0
  __shared__ bf16  xrbuf[128 * 16];      // [pixel][ci] conv1 result
  __shared__ float atbuf[128 * 16];      // [pixel][k] pre-softmax logits

  const int t   = threadIdx.x;
  const int img = blockIdx.x >> 7;
  const int h   = blockIdx.x & 127;
  const size_t xrow = (size_t)img * kC * kHW + (size_t)h * kW; // + c*HW + m
  const size_t gp0  = (size_t)img * kHW + (size_t)h * kW;      // global pixel

  // Stage x tile (128 px * 256 ch): b128 loads (16B/lane), transpose to LDS.
  // 8192 float4 groups; c = group>>5, m = (group&31)*4.
  for (int i = 0; i < 32; ++i) {
    int g4 = i * 256 + t;
    int c = g4 >> 5, m = (g4 & 31) << 2;
    float4 v = *(const float4*)&x[xrow + (size_t)c * kHW + m];
    xbuf[(m + 0) * XSTR + c] = f2bf(v.x);
    xbuf[(m + 1) * XSTR + c] = f2bf(v.y);
    xbuf[(m + 2) * XSTR + c] = f2bf(v.z);
    xbuf[(m + 3) * XSTR + c] = f2bf(v.w);
  }
  // Stage combined [w1 ; w2 ; zeros] weight block as bf16 (b128 loads).
  // 2048 float4 groups; r = group>>6, c = (group&63)*4.
  for (int i = 0; i < 8; ++i) {
    int g4 = i * 256 + t;
    int r = g4 >> 6, c = (g4 & 63) << 2;
    float4 v = make_float4(0.f, 0.f, 0.f, 0.f);
    if (r < 16)      v = *(const float4*)&w1[r * 256 + c];
    else if (r < 25) v = *(const float4*)&w2[(r - 16) * 256 + c];
    wbuf[r * WSTR + c + 0] = f2bf(v.x);
    wbuf[r * WSTR + c + 1] = f2bf(v.y);
    wbuf[r * WSTR + c + 2] = f2bf(v.z);
    wbuf[r * WSTR + c + 3] = f2bf(v.w);
  }
  __syncthreads();

  const int lane = t & 31;
  const int wv   = t >> 5;        // wave id: M-tile
  const int l    = lane & 15;     // row (A) / out-channel (B) within tile
  const int hb   = lane >> 4;     // half-wave selects K sub-block
  const int m0   = wv * 16;

  v8f acc1 = {}; v8f acc2 = {};
  #pragma unroll
  for (int ks = 0; ks < 8; ++ks) {
    const int kb = ks * 32;
    // A fragment: lanes 0-15 K={0..7,16..23}, lanes 16-31 K={8..15,24..31}
    const bf16* ap = &xbuf[(m0 + l) * XSTR + kb + hb * 8];
    v8bf alo = *(const v8bf*)ap;
    v8bf ahi = *(const v8bf*)(ap + 16);
    v16bf a;
    #pragma unroll
    for (int j = 0; j < 8; ++j) { a[j] = alo[j]; a[j + 8] = ahi[j]; }
    // B fragments (lane = output channel, VGPRs = K)
    const bf16* p1 = &wbuf[l * WSTR + kb + hb * 8];
    const bf16* p2 = &wbuf[(16 + l) * WSTR + kb + hb * 8];
    v8bf b1lo = *(const v8bf*)p1, b1hi = *(const v8bf*)(p1 + 16);
    v8bf b2lo = *(const v8bf*)p2, b2hi = *(const v8bf*)(p2 + 16);
    v16bf bb1, bb2;
    #pragma unroll
    for (int j = 0; j < 8; ++j) {
      bb1[j] = b1lo[j]; bb1[j + 8] = b1hi[j];
      bb2[j] = b2lo[j]; bb2[j + 8] = b2hi[j];
    }
    acc1 = __builtin_amdgcn_wmma_f32_16x16x32_bf16(false, a, false, bb1,
                                                   (short)0, acc1, false, false);
    acc2 = __builtin_amdgcn_wmma_f32_16x16x32_bf16(false, a, false, bb2,
                                                   (short)0, acc2, false, false);
  }

  // D layout: lane = n, VGPR r -> M = r (+8 for lanes 16-31)
  const float bias1 = b1[l];
  const float bias2 = (l < 9) ? b2[l] : 0.f;
  #pragma unroll
  for (int r = 0; r < 8; ++r) {
    const int m = m0 + r + hb * 8;
    float xr = acc1[r] + bias1; xr = xr > 0.f ? xr : 0.f;
    xrbuf[m * 16 + l] = f2bf(xr);
    float at = acc2[r] + bias2; at = at > 0.f ? at : 0.f;
    atbuf[m * 16 + l] = at;
  }
  __syncthreads();

  // Per-pixel softmax over the 9 window logits -> ws_attn (bf16, padded 16)
  if (t < 128) {
    float v[9]; float mx = -1e30f;
    #pragma unroll
    for (int k = 0; k < 9; ++k) { v[k] = atbuf[t * 16 + k]; mx = fmaxf(mx, v[k]); }
    float s = 0.f;
    #pragma unroll
    for (int k = 0; k < 9; ++k) { v[k] = __expf(v[k] - mx); s += v[k]; }
    const float inv = 1.f / s;
    v16bf o = {};
    #pragma unroll
    for (int k = 0; k < 9; ++k) o[k] = f2bf(v[k] * inv);
    *(v16bf*)&ws_attn[(gp0 + t) * 16] = o;
  }
  // Coalesced pixel-major xr store: 256 threads * b128
  *(v8bf*)&ws_xr[gp0 * 16 + t * 8] = *(const v8bf*)&xrbuf[t * 8];
}

// ------------------------------ Kernel 2 -----------------------------------
constexpr int VSTR  = 40;    // vbuf row stride  (80B: 20 dw, gcd(20,64)=4)
constexpr int W3STR = 24;    // w3buf row stride (48B: 12 dw, 16B aligned)
constexpr int OSTR  = 136;   // obuf row stride  (272B: 68 dw ≡ 4 mod 64)

__global__ __launch_bounds__(256, 1)
void k2_attn_conv3(const float* __restrict__ x,
                   const bf16* __restrict__ ws_xr,
                   const bf16* __restrict__ ws_attn,
                   const float* __restrict__ w3, const float* __restrict__ b3,
                   float* __restrict__ out)
{
  __shared__ bf16 vbuf[128 * VSTR];    // A operand: [pixel][K=32 (16 padded)]
  __shared__ bf16 w3buf[256 * W3STR];  // [co][ci]
  __shared__ bf16 obuf[256 * OSTR];    // [co][pixel] conv3 result

  const int t   = threadIdx.x;
  const int img = blockIdx.x >> 7;
  const int h   = blockIdx.x & 127;
  const size_t gp0 = (size_t)img * kHW + (size_t)h * kW;

  // Stage w3 (256 x 16) as bf16, 4x b128 loads per thread.
  {
    const float* wr = &w3[t * kCI];
    #pragma unroll
    for (int q = 0; q < 4; ++q) {
      float4 v = *(const float4*)&wr[q * 4];
      w3buf[t * W3STR + q * 4 + 0] = f2bf(v.x);
      w3buf[t * W3STR + q * 4 + 1] = f2bf(v.y);
      w3buf[t * W3STR + q * 4 + 2] = f2bf(v.z);
      w3buf[t * W3STR + q * 4 + 3] = f2bf(v.w);
    }
  }

  // 3x3 attention-weighted gather: 2 threads / pixel (ci halves 0-7 / 8-15).
  {
    const int m  = t & 127;
    const int h2 = t >> 7;
    v16bf at = *(const v16bf*)&ws_attn[(gp0 + m) * 16];
    float vacc[8];
    #pragma unroll
    for (int j = 0; j < 8; ++j) vacc[j] = 0.f;
    #pragma unroll
    for (int di = 0; di < 3; ++di) {
      const int hh = h + di - 1;
      #pragma unroll
      for (int dj = 0; dj < 3; ++dj) {
        const int ww = m + dj - 1;
        const float ak = bf2f(at[di * 3 + dj]);
        if (hh >= 0 && hh < kH && ww >= 0 && ww < kW) {   // zero padding
          const size_t pn = (size_t)img * kHW + (size_t)hh * kW + ww;
          v8bf xrv = *(const v8bf*)&ws_xr[pn * 16 + h2 * 8];
          #pragma unroll
          for (int j = 0; j < 8; ++j) vacc[j] += ak * bf2f(xrv[j]);
        }
      }
    }
    #pragma unroll
    for (int j = 0; j < 8; ++j) {
      vbuf[m * VSTR + h2 * 8 + j]      = f2bf(vacc[j]);
      vbuf[m * VSTR + 16 + h2 * 8 + j] = f2bf(0.f);       // K 16..31 pad
    }
  }
  __syncthreads();

  const int lane = t & 31;
  const int wv   = t >> 5;
  const int l    = lane & 15;
  const int hb   = lane >> 4;
  const int m0   = wv * 16;

  // One A fragment per wave (K = 32, upper half zero).
  const bf16* ap = &vbuf[(m0 + l) * VSTR + hb * 8];
  v8bf alo = *(const v8bf*)ap;
  v8bf ahi = *(const v8bf*)(ap + 16);
  v16bf a;
  #pragma unroll
  for (int j = 0; j < 8; ++j) { a[j] = alo[j]; a[j + 8] = ahi[j]; }

  // Sweep 16 N-tiles -> all 256 output channels.
  #pragma unroll
  for (int nt = 0; nt < 16; ++nt) {
    const int co = nt * 16 + l;
    v8bf blo = *(const v8bf*)&w3buf[co * W3STR + hb * 8];
    v16bf b = {};
    #pragma unroll
    for (int j = 0; j < 8; ++j) b[j] = blo[j];            // K 16..31 = 0
    v8f acc = {};
    acc = __builtin_amdgcn_wmma_f32_16x16x32_bf16(false, a, false, b,
                                                  (short)0, acc, false, false);
    const float bias = b3[co];
    #pragma unroll
    for (int r = 0; r < 8; ++r) {
      float d = acc[r] + bias; d = d > 0.f ? d : 0.f;     // relu
      obuf[co * OSTR + (m0 + r + hb * 8)] = f2bf(d);
    }
  }
  __syncthreads();

  // Residual add + store, b128 both ways: out = x + relu(conv3(...)).
  // 8192 float4 groups; c = group>>5, m = (group&31)*4.
  const size_t xrow = (size_t)img * kC * kHW + (size_t)h * kW;
  for (int i = 0; i < 32; ++i) {
    int g4 = i * 256 + t;
    int c = g4 >> 5, m = (g4 & 31) << 2;
    const size_t g = xrow + (size_t)c * kHW + m;
    float4 xv = *(const float4*)&x[g];
    float4 ov;
    ov.x = xv.x + bf2f(obuf[c * OSTR + m + 0]);
    ov.y = xv.y + bf2f(obuf[c * OSTR + m + 1]);
    ov.z = xv.z + bf2f(obuf[c * OSTR + m + 2]);
    ov.w = xv.w + bf2f(obuf[c * OSTR + m + 3]);
    *(float4*)&out[g] = ov;
  }
}

// ------------------------------ Launcher -----------------------------------
extern "C" void kernel_launch(void* const* d_in, const int* in_sizes, int n_in,
                              void* d_out, int out_size, void* d_ws, size_t ws_size,
                              hipStream_t stream) {
  const float* x  = (const float*)d_in[0];
  const float* w1 = (const float*)d_in[1];
  const float* b1 = (const float*)d_in[2];
  const float* w2 = (const float*)d_in[3];
  const float* b2 = (const float*)d_in[4];
  const float* w3 = (const float*)d_in[5];
  const float* b3 = (const float*)d_in[6];
  float* out = (float*)d_out;

  const size_t npix = (size_t)kN * kHW;
  bf16* ws_xr   = (bf16*)d_ws;                 // npix * 16 bf16 = 8.4 MB
  bf16* ws_attn = ws_xr + npix * 16;           // npix * 16 bf16 = 8.4 MB

  dim3 grid(kN * kH);   // one block per image row
  k1_conv12<<<grid, 256, 0, stream>>>(x, w1, b1, w2, b2, ws_xr, ws_attn);
  k2_attn_conv3<<<grid, 256, 0, stream>>>(x, ws_xr, ws_attn, w3, b3, out);
}
